// RNNEncoder3D_71708773974125
// MI455X (gfx1250) — matrix operand order repaired and checked by reading, not compiled, not run on hardware
//
#include <hip/hip_runtime.h>
#include <math.h>

typedef __attribute__((ext_vector_type(16))) __bf16 v16bf;
typedef __attribute__((ext_vector_type(8)))  float  v8f;

#define H_      768
#define THREE_H 2304
#define B_      16
#define T_      1024
#define ROWS    16384        // B*T
#define MT_     1024         // ROWS/16
#define KT_     24           // 768/32 k-tiles
#define NT_G    144          // 2304/16 n-tiles (gates)
#define NT_FC   32           // 512/16 n-tiles (fc)
#define PANEL   512          // 32 lanes * 16 elems per WMMA operand panel
#define NWG_SCAN 48          // 768/16 column groups
#define GEMM_MB 2            // M-tiles per wave (register blocking)
#define GEMM_NB 4            // N-tiles per wave

__device__ __forceinline__ unsigned short f2bf(float f) {
  unsigned u = __float_as_uint(f);
  unsigned r = 0x7FFFu + ((u >> 16) & 1u);   // round-to-nearest-even
  return (unsigned short)((u + r) >> 16);
}
__device__ __forceinline__ float sigmoidf_(float x) { return 1.0f / (1.0f + expf(-x)); }

// ---------------------------------------------------------------------------
// Pack fp32 row-major [R x K] matrix into WMMA A-operand panels (16x32 bf16).
// Layout per ISA 7.12.2: lane = m + 16*((k>>3)&1); slot = (k&7) + 8*((k>>4)&1)
// ---------------------------------------------------------------------------
__global__ void pack_a_kernel(const float* __restrict__ in, unsigned short* __restrict__ out,
                              int rtiles, int ktiles, int K) {
  long long tid = (long long)blockIdx.x * blockDim.x + threadIdx.x;
  long long total = (long long)rtiles * ktiles * PANEL;
  if (tid >= total) return;
  int within = (int)(tid & (PANEL - 1));
  long long panel = tid >> 9;
  int lane = within >> 4, slot = within & 15;
  int kt = (int)(panel % ktiles);
  int mt = (int)(panel / ktiles);
  int m = lane & 15, hi = lane >> 4;
  int k = kt * 32 + ((slot < 8) ? slot : slot + 8) + 8 * hi;
  int row = mt * 16 + m;
  out[tid] = f2bf(in[(long long)row * K + k]);
}

// ---------------------------------------------------------------------------
// Pack fp32 weight [N x K] into WMMA B-operand panels for B = W^T (K x N).
// Per ISA: lane = (n&15) + 16*((k>>3)&1), slot = (k&7) + 8*((k>>4)&1) within a 32-K tile.
// ---------------------------------------------------------------------------
__global__ void pack_b_kernel(const float* __restrict__ w, unsigned short* __restrict__ out,
                              int ntiles, int ktiles, int K) {
  long long tid = (long long)blockIdx.x * blockDim.x + threadIdx.x;
  long long total = (long long)ntiles * ktiles * PANEL;
  if (tid >= total) return;
  int within = (int)(tid & (PANEL - 1));
  long long panel = tid >> 9;
  int lane = within >> 4, slot = within & 15;
  int kt = (int)(panel % ktiles);
  int nt = (int)(panel / ktiles);
  int n = nt * 16 + (lane & 15);
  int k = kt * 32 + ((slot < 8) ? slot : slot + 8) + 8 * (lane >> 4);
  out[tid] = f2bf(w[(long long)n * K + k]);
}

// ---------------------------------------------------------------------------
// Dense bf16 WMMA GEMM with 2x4 register blocking per wave:
// each wave computes a 32(M) x 64(N) macro-tile as 8 accumulators, so every
// loaded 1KB operand panel feeds 4 (A) or 2 (B) WMMAs (~21 FLOP/byte from L2).
// out[row, col] = gemm + bias[col], optional exact-GELU epilogue.
// ---------------------------------------------------------------------------
__global__ void __launch_bounds__(128) wmma_gemm_kernel(
    const unsigned short* __restrict__ Ap, const unsigned short* __restrict__ Bp,
    const float* __restrict__ bias, float* __restrict__ out,
    int ktiles, int ldOut, int applyGelu) {
  const int lane = threadIdx.x & 31;
  const int wave = threadIdx.x >> 5;
  const int ntBase = (blockIdx.x * 4 + wave) * GEMM_NB;
  const int mtBase = blockIdx.y * GEMM_MB;

  const unsigned short* aPtr[GEMM_MB];
  const unsigned short* bPtr[GEMM_NB];
#pragma unroll
  for (int mi = 0; mi < GEMM_MB; ++mi)
    aPtr[mi] = Ap + ((size_t)(mtBase + mi) * ktiles) * PANEL + lane * 16;
#pragma unroll
  for (int ni = 0; ni < GEMM_NB; ++ni)
    bPtr[ni] = Bp + ((size_t)(ntBase + ni) * ktiles) * PANEL + lane * 16;

  v8f acc[GEMM_MB][GEMM_NB];
#pragma unroll
  for (int mi = 0; mi < GEMM_MB; ++mi)
#pragma unroll
    for (int ni = 0; ni < GEMM_NB; ++ni) acc[mi][ni] = (v8f){};

  for (int kt = 0; kt < ktiles; ++kt) {
    v16bf aOp[GEMM_MB], bOp[GEMM_NB];
#pragma unroll
    for (int mi = 0; mi < GEMM_MB; ++mi) aOp[mi] = *(const v16bf*)(aPtr[mi] + kt * PANEL);
#pragma unroll
    for (int ni = 0; ni < GEMM_NB; ++ni) bOp[ni] = *(const v16bf*)(bPtr[ni] + kt * PANEL);
#pragma unroll
    for (int mi = 0; mi < GEMM_MB; ++mi)
#pragma unroll
      for (int ni = 0; ni < GEMM_NB; ++ni)
        acc[mi][ni] = __builtin_amdgcn_wmma_f32_16x16x32_bf16(
            false, aOp[mi], false, bOp[ni], (short)0, acc[mi][ni], false, false);
  }

  const int n = lane & 15;
  const int mbase = (lane >> 4) * 8;
#pragma unroll
  for (int ni = 0; ni < GEMM_NB; ++ni) {
    const int col = (ntBase + ni) * 16 + n;
    const float bc = bias[col];
#pragma unroll
    for (int mi = 0; mi < GEMM_MB; ++mi) {
#pragma unroll
      for (int v = 0; v < 8; ++v) {
        const int row = (mtBase + mi) * 16 + mbase + v;
        float val = acc[mi][ni][v] + bc;
        if (applyGelu) val = 0.5f * val * (1.0f + erff(val * 0.70710678118654752f));
        out[(size_t)row * ldOut + col] = val;
      }
    }
  }
}

// ---------------------------------------------------------------------------
// Re-init scan state each call (workspace is poisoned): barrier counter,
// f32 h double buffer, packed-bf16 h double buffer (h0 = 0).
// ---------------------------------------------------------------------------
__global__ void scan_init_kernel(float* __restrict__ hF, unsigned short* __restrict__ hP,
                                 unsigned int* __restrict__ ctr) {
  int tid = blockIdx.x * blockDim.x + threadIdx.x;
  int stride = gridDim.x * blockDim.x;
  if (tid == 0) ctr[0] = 0u;
  for (int i = tid; i < 2 * B_ * H_; i += stride) hF[i] = 0.0f;
  for (int i = tid; i < 2 * KT_ * PANEL; i += stride) hP[i] = 0;
}

// ---------------------------------------------------------------------------
// Persistent GRU scan: 48 WGs x 1 wave. WG `wg` owns h columns [16*wg, 16*wg+16).
// Its 72KB w_hh slice (3 gates x 24 k-panels) is staged ONCE into LDS; the
// per-step critical path is then 24 global h-panel loads + 72 LDS-fed
// v_wmma_f32_16x16x32_bf16, fused gate math on the f32 accumulators, and a
// device-scope atomic barrier (monotonic counter) per step.
// ---------------------------------------------------------------------------
__global__ void __launch_bounds__(32, 1) gru_scan_kernel(
    const float* __restrict__ gates,       // [B,T,3H] f32
    const unsigned short* __restrict__ Wp, // w_hh B-panels [144 ntiles][24 kt][512]
    const float* __restrict__ bhh,         // [3H]
    float* __restrict__ hF,                // [2][B,H] f32
    unsigned short* __restrict__ hP,       // [2][24][512] packed bf16 A panels
    unsigned short* __restrict__ outP,     // [MT][24][512] packed A panels (layer out)
    unsigned int* __restrict__ ctr, int T) {
  extern __shared__ unsigned short ldsW[]; // 3 * KT_ * PANEL bf16 = 72KB

  const int wg   = blockIdx.x;      // 0..47
  const int lane = threadIdx.x;     // 0..31
  const int n    = lane & 15;
  const int col  = wg * 16 + n;     // output h column owned by this lane
  const int mbase = (lane >> 4) * 8;

  // one-time stage of this WG's w_hh slice into LDS (L2 -> LDS, 72KB)
#pragma unroll
  for (int g = 0; g < 3; ++g)
    for (int kt = 0; kt < KT_; ++kt) {
      *(v16bf*)(ldsW + (g * KT_ + kt) * PANEL + lane * 16) =
          *(const v16bf*)(Wp + ((size_t)(g * NWG_SCAN + wg) * KT_ + kt) * PANEL + lane * 16);
    }
  __syncthreads();

  const float bhr = bhh[col];
  const float bhz = bhh[H_ + col];
  const float bhn = bhh[2 * H_ + col];

  // packed-layout coordinates of (m, col) within its 16x32 k-tile
  const int ktc = col >> 5;
  const int kk  = col & 31;
  const int pl  = 16 * ((kk >> 3) & 1);             // lane offset (+16 half)
  const int ps  = (kk & 7) + 8 * ((kk >> 4) & 1);   // slot

  const unsigned short* lR = ldsW + (0 * KT_) * PANEL + lane * 16;
  const unsigned short* lZ = ldsW + (1 * KT_) * PANEL + lane * 16;
  const unsigned short* lN = ldsW + (2 * KT_) * PANEL + lane * 16;

  for (int t = 0; t < T; ++t) {
    const int cur = t & 1, nxt = cur ^ 1;
    const unsigned short* ap = hP + cur * (KT_ * PANEL) + lane * 16;

    v8f aR = {}, aZ = {}, aN = {};
    for (int kt = 0; kt < KT_; ++kt) {
      v16bf a  = *(const v16bf*)(ap + kt * PANEL);
      v16bf wr = *(const v16bf*)(lR + kt * PANEL);
      v16bf wz = *(const v16bf*)(lZ + kt * PANEL);
      v16bf wn = *(const v16bf*)(lN + kt * PANEL);
      aR = __builtin_amdgcn_wmma_f32_16x16x32_bf16(false, a, false, wr, (short)0, aR, false, false);
      aZ = __builtin_amdgcn_wmma_f32_16x16x32_bf16(false, a, false, wz, (short)0, aZ, false, false);
      aN = __builtin_amdgcn_wmma_f32_16x16x32_bf16(false, a, false, wn, (short)0, aN, false, false);
    }

#pragma unroll
    for (int v = 0; v < 8; ++v) {
      const int m = mbase + v;                     // batch index
      const float* g = gates + ((size_t)(m * T + t)) * THREE_H;
      float r  = sigmoidf_(g[col]          + aR[v] + bhr);
      float z  = sigmoidf_(g[H_ + col]     + aZ[v] + bhz);
      float nn = tanhf    (g[2 * H_ + col] + r * (aN[v] + bhn));
      float hp = hF[cur * (B_ * H_) + m * H_ + col];
      float hn = (1.0f - z) * nn + z * hp;
      hF[nxt * (B_ * H_) + m * H_ + col] = hn;
      unsigned short hb = f2bf(hn);
      hP[nxt * (KT_ * PANEL) + ktc * PANEL + (m + pl) * 16 + ps] = hb;
      const int row = m * T + t;                   // row in [B*T, H] layer output
      outP[((size_t)(row >> 4) * KT_ + ktc) * PANEL + ((row & 15) + pl) * 16 + ps] = hb;
    }

    __threadfence();                               // release h/out writes (agent scope)
    if (lane == 0)
      __hip_atomic_fetch_add(ctr, 1u, __ATOMIC_RELEASE, __HIP_MEMORY_SCOPE_AGENT);
    const unsigned target = (unsigned)NWG_SCAN * (unsigned)(t + 1);
    while (__hip_atomic_load(ctr, __ATOMIC_ACQUIRE, __HIP_MEMORY_SCOPE_AGENT) < target)
      __builtin_amdgcn_s_sleep(1);
  }
}

// ---------------------------------------------------------------------------
extern "C" void kernel_launch(void* const* d_in, const int* in_sizes, int n_in,
                              void* d_out, int out_size, void* d_ws, size_t ws_size,
                              hipStream_t stream) {
  (void)in_sizes; (void)n_in; (void)out_size; (void)ws_size;
  const float* x    = (const float*)d_in[0];   // [16,1024,768]
  const float* w_ih = (const float*)d_in[1];   // [2,2304,768]
  const float* w_hh = (const float*)d_in[2];   // [2,2304,768]
  const float* b_ih = (const float*)d_in[3];   // [2,2304]
  const float* b_hh = (const float*)d_in[4];   // [2,2304]
  const float* w_fc = (const float*)d_in[5];   // [512,768]
  const float* b_fc = (const float*)d_in[6];   // [512]
  float* out = (float*)d_out;                  // [16,1024,512]

  char* ws = (char*)d_ws;
  size_t off = 0;
  auto wsAlloc = [&](size_t bytes) -> void* {
    void* p = ws + off;
    off = (off + bytes + 255) & ~(size_t)255;
    return p;
  };

  const size_t aElems = (size_t)MT_ * KT_ * PANEL;     // packed A panel elems for [16384x768]
  const size_t wElems = (size_t)NT_G * KT_ * PANEL;    // packed B panel elems for [2304x768]

  unsigned short* bufA  = (unsigned short*)wsAlloc(aElems * 2);
  unsigned short* bufB  = (unsigned short*)wsAlloc(aElems * 2);
  float*          gates = (float*)wsAlloc((size_t)ROWS * THREE_H * 4);
  unsigned short* wih_p = (unsigned short*)wsAlloc(2 * wElems * 2);
  unsigned short* whh_p = (unsigned short*)wsAlloc(2 * wElems * 2);
  unsigned short* wfc_p = (unsigned short*)wsAlloc((size_t)NT_FC * KT_ * PANEL * 2);
  float*          hF    = (float*)wsAlloc(2 * B_ * H_ * 4);
  unsigned short* hP    = (unsigned short*)wsAlloc(2 * KT_ * PANEL * 2);
  unsigned int*   ctr   = (unsigned int*)wsAlloc(256);

  // ---- pack inputs / weights to bf16 WMMA panel layouts ----
  {
    int thr = 256;
    long long blkA = ((long long)aElems + thr - 1) / thr;
    pack_a_kernel<<<(unsigned)blkA, thr, 0, stream>>>(x, bufA, MT_, KT_, H_);
    long long blkW = ((long long)wElems + thr - 1) / thr;
    for (int l = 0; l < 2; ++l) {
      pack_b_kernel<<<(unsigned)blkW, thr, 0, stream>>>(w_ih + (size_t)l * THREE_H * H_,
                                                        wih_p + (size_t)l * wElems, NT_G, KT_, H_);
      pack_b_kernel<<<(unsigned)blkW, thr, 0, stream>>>(w_hh + (size_t)l * THREE_H * H_,
                                                        whh_p + (size_t)l * wElems, NT_G, KT_, H_);
    }
    long long fcElems = (long long)NT_FC * KT_ * PANEL;
    pack_b_kernel<<<(unsigned)((fcElems + thr - 1) / thr), thr, 0, stream>>>(w_fc, wfc_p, NT_FC, KT_, H_);
  }

  // waves per block = 4; each wave covers 4 n-tiles and 2 m-tiles
  dim3 ggates(NT_G / (4 * GEMM_NB), MT_ / GEMM_MB);   // (9, 512)
  dim3 gfc(NT_FC / (4 * GEMM_NB), MT_ / GEMM_MB);     // (2, 512)
  const size_t scanLds = (size_t)3 * KT_ * PANEL * sizeof(unsigned short);  // 73728 B

  // ---- layer 0 ----
  wmma_gemm_kernel<<<ggates, 128, 0, stream>>>(bufA, wih_p, b_ih, gates, KT_, THREE_H, 0);
  scan_init_kernel<<<96, 256, 0, stream>>>(hF, hP, ctr);
  gru_scan_kernel<<<NWG_SCAN, 32, scanLds, stream>>>(gates, whh_p, b_hh, hF, hP, bufB, ctr, T_);

  // ---- layer 1 ----
  wmma_gemm_kernel<<<ggates, 128, 0, stream>>>(bufB, wih_p + wElems, b_ih + THREE_H,
                                               gates, KT_, THREE_H, 0);
  scan_init_kernel<<<96, 256, 0, stream>>>(hF, hP, ctr);
  gru_scan_kernel<<<NWG_SCAN, 32, scanLds, stream>>>(gates, whh_p + wElems, b_hh + THREE_H,
                                                     hF, hP, bufA, ctr, T_);

  // ---- FC + exact GELU ----
  wmma_gemm_kernel<<<gfc, 128, 0, stream>>>(bufA, wfc_p, b_fc, out, KT_, 512, 1);
}
